// LTAE_65721589563732
// MI455X (gfx1250) — matrix operand order repaired and checked by reading, not compiled, not run on hardware
//
#include <hip/hip_runtime.h>
#include <hip/hip_bf16.h>

// ---------------------------------------------------------------------------
// LTAE forward for MI455X (gfx1250, wave32, WMMA).
// Big GEMMs (286720x128 @ 128x256 and 286720x256 @ 256x128, ~38 GFLOP total)
// run on v_wmma_f32_16x16x32_f16 with f32 accumulation. Everything else is
// HBM-bound elementwise/reduction work, fused to minimize passes over the
// ~0.3-0.7 GB intermediates (23.3 TB/s HBM -> ~0.1-0.2 ms total traffic).
// ---------------------------------------------------------------------------

#define BB   4096
#define SS   70
#define DIN  10
#define DM   256
#define NH   16
#define DKK  8
#define BS   (BB * SS)          // 286720, multiple of 16

typedef __attribute__((ext_vector_type(16))) _Float16 v16h;
typedef __attribute__((ext_vector_type(8)))  float    v8f;

// ---------------- WMMA fragment loaders (ISA 7.12.2 layouts) ----------------
// A 16x32 f16: lanes 0-15 -> M=lane, K = e (e<8) / e+8 (e>=8);
//              lanes 16-31 -> same M, K shifted +8. K pairs are contiguous.
__device__ __forceinline__ v16h load_fragA(const float* __restrict__ A, int lda,
                                           int m0, int k0, int lane) {
  const int m   = m0 + (lane & 15);
  const int klo = (lane & 16) ? 8 : 0;
  const float* row = A + (size_t)m * lda + (k0 + klo);
  v16h a;
#pragma unroll
  for (int t = 0; t < 8; ++t) {
    const int koff = (t < 4) ? (2 * t) : (2 * t + 8);
    float2 f = *(const float2*)(row + koff);
    a[2 * t]     = (_Float16)f.x;
    a[2 * t + 1] = (_Float16)f.y;
  }
  return a;
}

// B 32x16 f16 from weights stored [N,K] row-major (y = x @ W^T):
// lanes 0-15 -> N=lane, K = e (e<8) / e+8 (e>=8); lanes 16-31 -> K shifted +8. K pairs contiguous.
__device__ __forceinline__ v16h load_fragB(const float* __restrict__ W, int ldw,
                                           int n0, int k0, int lane) {
  const int n   = n0 + (lane & 15);
  const int klo = (lane & 16) ? 8 : 0;
  const float* row = W + (size_t)n * ldw + (k0 + klo);
  v16h b;
#pragma unroll
  for (int t = 0; t < 8; ++t) {
    float2 f = *(const float2*)(row + ((t < 4) ? (2 * t) : (2 * t + 8)));
    b[2 * t]     = (_Float16)f.x;
    b[2 * t + 1] = (_Float16)f.y;
  }
  return b;
}

// C = A[MxK] @ W[NxK]^T + bias, one 16x16 tile per wave, 8 waves per block.
template <int K>
__global__ __launch_bounds__(256)
void wmma_gemm_bias_kernel(const float* __restrict__ A, const float* __restrict__ W,
                           const float* __restrict__ bias, float* __restrict__ C,
                           int M, int N) {
  const int lane = threadIdx.x & 31;
  const int wave = threadIdx.x >> 5;
  const int ntiles = N >> 4;
  const long tile = (long)blockIdx.x * 8 + wave;
  const long mt = tile / ntiles;
  const int  nt = (int)(tile % ntiles);
  if (mt * 16 >= M) return;
  const int m0 = (int)(mt * 16);
  const int n0 = nt * 16;

  v8f acc = {};
#pragma unroll
  for (int k0 = 0; k0 < K; k0 += 32) {
    v16h a = load_fragA(A, K, m0, k0, lane);
    v16h b = load_fragB(W, K, n0, k0, lane);
    acc = __builtin_amdgcn_wmma_f32_16x16x32_f16(
        /*neg_a=*/false, a, /*neg_b=*/false, b,
        /*c_mod=*/(short)0, acc, /*reuse_a=*/false, /*reuse_b=*/false);
  }
  // C/D layout: VGPR r -> M = m0 + r (+8 for lanes 16-31), N = n0 + (lane&15)
  const int n = n0 + (lane & 15);
  const int mb = m0 + ((lane & 16) ? 8 : 0);
  const float bn = bias ? bias[n] : 0.0f;
#pragma unroll
  for (int r = 0; r < 8; ++r)
    C[(size_t)(mb + r) * N + n] = acc[r] + bn;
}

// ------------- scalar fused linear: Y = relu(X*scale+shift) @ W^T + b -------
template <int DINc, int DOUTc>
__global__ __launch_bounds__(256)
void affine_relu_linear_kernel(const float* __restrict__ X,
                               const float* __restrict__ scale,
                               const float* __restrict__ shift,
                               const float* __restrict__ W,
                               const float* __restrict__ bias,
                               float* __restrict__ Y, long R) {
  __shared__ float sW[DOUTc * DINc];
  __shared__ float sb[DOUTc];
  __shared__ float ssc[DINc], ssh[DINc];
  for (int i = threadIdx.x; i < DOUTc * DINc; i += blockDim.x) sW[i] = W[i];
  for (int i = threadIdx.x; i < DOUTc; i += blockDim.x) sb[i] = bias[i];
  if (scale)
    for (int i = threadIdx.x; i < DINc; i += blockDim.x) { ssc[i] = scale[i]; ssh[i] = shift[i]; }
  __syncthreads();

  const long r = (long)blockIdx.x * blockDim.x + threadIdx.x;
  if (r >= R) return;
  float xin[DINc];
  const float* xr = X + (size_t)r * DINc;
#pragma unroll
  for (int i = 0; i < DINc; ++i) {
    float v = xr[i];
    if (scale) v = fmaxf(v * ssc[i] + ssh[i], 0.0f);
    xin[i] = v;
  }
  float* yr = Y + (size_t)r * DOUTc;
  for (int j = 0; j < DOUTc; ++j) {
    float acc = sb[j];
    const float* wj = &sW[j * DINc];
#pragma unroll
    for (int i = 0; i < DINc; ++i) acc = fmaf(xin[i], wj[i], acc);
    yr[j] = acc;
  }
}

// --------- BatchNorm column stats folded to affine: scale/shift -------------
__global__ __launch_bounds__(256)
void bn_stats_kernel(const float* __restrict__ Z, long R, int Cn,
                     const float* __restrict__ g, const float* __restrict__ be,
                     float* __restrict__ scale, float* __restrict__ shift) {
  const int c = blockIdx.x;
  __shared__ float s1[256], s2[256];
  float sum = 0.0f, sq = 0.0f;
  for (long r = threadIdx.x; r < R; r += 256) {
    float v = Z[(size_t)r * Cn + c];
    sum += v; sq += v * v;
  }
  s1[threadIdx.x] = sum; s2[threadIdx.x] = sq;
  __syncthreads();
  for (int off = 128; off; off >>= 1) {
    if (threadIdx.x < off) { s1[threadIdx.x] += s1[threadIdx.x + off];
                             s2[threadIdx.x] += s2[threadIdx.x + off]; }
    __syncthreads();
  }
  if (threadIdx.x == 0) {
    float m = s1[0] / (float)R;
    float v = s2[0] / (float)R - m * m;
    float a = g[c] * rsqrtf(v + 1e-5f);
    scale[c] = a;
    shift[c] = be[c] - m * a;
  }
}

// --------- fused BN-affine -> ReLU -> LayerNorm(128) ------------------------
__global__ __launch_bounds__(128)
void bn_relu_ln_kernel(const float* __restrict__ Z, const float* __restrict__ scale,
                       const float* __restrict__ shift, const float* __restrict__ g,
                       const float* __restrict__ bb, float* __restrict__ out, long R) {
  const long r = blockIdx.x;
  const int t = threadIdx.x;
  float v = Z[(size_t)r * 128 + t];
  v = fmaxf(v * scale[t] + shift[t], 0.0f);
  __shared__ float s1[128], s2[128];
  s1[t] = v; s2[t] = v * v;
  __syncthreads();
  for (int off = 64; off; off >>= 1) {
    if (t < off) { s1[t] += s1[t + off]; s2[t] += s2[t + off]; }
    __syncthreads();
  }
  const float m = s1[0] * (1.0f / 128.0f);
  const float var = s2[0] * (1.0f / 128.0f) - m * m;
  const float rs = rsqrtf(var + 1e-5f);
  out[(size_t)r * 128 + t] = (v - m) * rs * g[t] + bb[t];
}

// --------- per-sample LN((DM,S)) stats over S*DM elements -------------------
__global__ __launch_bounds__(256)
void sample_stats_kernel(const float* __restrict__ Hc, float* __restrict__ mean,
                         float* __restrict__ rstd) {
  const int b = blockIdx.x;
  const float* p = Hc + (size_t)b * SS * DM;
  __shared__ float s1[256], s2[256];
  float sum = 0.0f, sq = 0.0f;
  for (int i = threadIdx.x; i < SS * DM; i += 256) {
    float v = p[i];
    sum += v; sq += v * v;
  }
  s1[threadIdx.x] = sum; s2[threadIdx.x] = sq;
  __syncthreads();
  for (int off = 128; off; off >>= 1) {
    if (threadIdx.x < off) { s1[threadIdx.x] += s1[threadIdx.x + off];
                             s2[threadIdx.x] += s2[threadIdx.x + off]; }
    __syncthreads();
  }
  if (threadIdx.x == 0) {
    const float Nn = (float)(SS * DM);
    float m = s1[0] / Nn;
    float v = s2[0] / Nn - m * m;
    mean[b] = m;
    rstd[b] = rsqrtf(v + 1e-5f);
  }
}

// --------- normalize + ln2 affine + positional encoding (in-place) ---------
__global__ __launch_bounds__(256)
void encode_kernel(float* __restrict__ Hc, const float* __restrict__ mean,
                   const float* __restrict__ rstd, const float* __restrict__ ln2g,
                   const float* __restrict__ ln2b, const int* __restrict__ doy,
                   const int* __restrict__ mask, const float* __restrict__ pos) {
  const long row = blockIdx.x;          // BS rows
  const int d = threadIdx.x;            // DM channels
  const int b = (int)(row / SS), s = (int)(row % SS);
  int dd = mask[b * SS + s] ? 0 : (doy[b * SS + s] - doy[b * SS]);
  dd = (dd < 0) ? 0 : ((dd > 366) ? 366 : dd);
  const size_t idx = (size_t)row * DM + d;
  float v = (Hc[idx] - mean[b]) * rstd[b];
  v = v * ln2g[d * SS + s] + ln2b[d * SS + s];   // ln2_g.T / ln2_b.T
  Hc[idx] = v + pos[(size_t)dd * DM + d];
}

// --------- master-query attention: one wave32 per head ----------------------
__global__ __launch_bounds__(512)
void attention_kernel(const float* __restrict__ kbuf, const float* __restrict__ enc,
                      const float* __restrict__ Q, float* __restrict__ obuf) {
  const int b = blockIdx.x;
  const int lane = threadIdx.x & 31;
  const int h = threadIdx.x >> 5;       // 16 waves = 16 heads
  __shared__ float sco[NH][SS];
  __shared__ float Qs[NH * DKK];
  if (threadIdx.x < NH * DKK) Qs[threadIdx.x] = Q[threadIdx.x];
  __syncthreads();

  float smax = -1e30f;
  for (int s = lane; s < SS; s += 32) {
    const float* kp = kbuf + ((size_t)(b * SS + s)) * (NH * DKK) + h * DKK;
    float sc = 0.0f;
#pragma unroll
    for (int d = 0; d < DKK; ++d) sc = fmaf(Qs[h * DKK + d], kp[d], sc);
    sc *= 0.35355339059327373f;         // 1/sqrt(DK)
    sco[h][s] = sc;
    smax = fmaxf(smax, sc);
  }
  for (int off = 16; off; off >>= 1) smax = fmaxf(smax, __shfl_xor(smax, off, 32));
  float ssum = 0.0f;
  for (int s = lane; s < SS; s += 32) {
    float e = __expf(sco[h][s] - smax);
    sco[h][s] = e;
    ssum += e;
  }
  for (int off = 16; off; off >>= 1) ssum += __shfl_xor(ssum, off, 32);
  const float inv = 1.0f / ssum;
  __syncthreads();

  if (lane < 16) {                      // value head width DM/NH = 16
    float acc = 0.0f;
    for (int s = 0; s < SS; ++s)
      acc = fmaf(sco[h][s], enc[((size_t)(b * SS + s)) * DM + h * 16 + lane], acc);
    obuf[(size_t)b * DM + h * 16 + lane] = acc * inv;
  }
}

// ---------------------------------------------------------------------------
extern "C" void kernel_launch(void* const* d_in, const int* in_sizes, int n_in,
                              void* d_out, int out_size, void* d_ws, size_t ws_size,
                              hipStream_t stream) {
  (void)in_sizes; (void)n_in; (void)out_size; (void)ws_size;
  const float* x     = (const float*)d_in[0];
  const int*   maskp = (const int*)  d_in[1];
  const int*   doyp  = (const int*)  d_in[2];
  const float* w1 = (const float*)d_in[3];  const float* b1  = (const float*)d_in[4];
  const float* g1 = (const float*)d_in[5];  const float* be1 = (const float*)d_in[6];
  const float* w2 = (const float*)d_in[7];  const float* b2  = (const float*)d_in[8];
  const float* g2 = (const float*)d_in[9];  const float* be2 = (const float*)d_in[10];
  const float* w3 = (const float*)d_in[11]; const float* b3  = (const float*)d_in[12];
  const float* g3 = (const float*)d_in[13]; const float* be3 = (const float*)d_in[14];
  const float* ln_in_g = (const float*)d_in[15]; const float* ln_in_b = (const float*)d_in[16];
  const float* conv_w  = (const float*)d_in[17]; const float* conv_b  = (const float*)d_in[18];
  const float* ln2_g   = (const float*)d_in[19]; const float* ln2_b   = (const float*)d_in[20];
  const float* Q       = (const float*)d_in[21];
  const float* wk = (const float*)d_in[22]; const float* bk = (const float*)d_in[23];
  const float* w_m2 = (const float*)d_in[24]; const float* b_m2 = (const float*)d_in[25];
  const float* g_m2 = (const float*)d_in[26]; const float* be_m2 = (const float*)d_in[27];
  const float* out_g = (const float*)d_in[28]; const float* out_b = (const float*)d_in[29];
  const float* wd1 = (const float*)d_in[30]; const float* bd1  = (const float*)d_in[31];
  const float* gd1 = (const float*)d_in[32]; const float* bed1 = (const float*)d_in[33];
  const float* wd2 = (const float*)d_in[34]; const float* bd2  = (const float*)d_in[35];
  const float* gd2 = (const float*)d_in[36]; const float* bed2 = (const float*)d_in[37];
  const float* wd3 = (const float*)d_in[38]; const float* bd3  = (const float*)d_in[39];
  const float* pos = (const float*)d_in[40];

  float* ws = (float*)d_ws;
  size_t off = 0;
  auto alloc = [&](size_t n) { float* p = ws + off; off += n; return p; };

  // fold-to-affine BN params
  float* sc1 = alloc(32);  float* sh1 = alloc(32);
  float* sc2 = alloc(64);  float* sh2 = alloc(64);
  float* sc3 = alloc(128); float* sh3 = alloc(128);
  float* scm2 = alloc(128); float* shm2 = alloc(128);
  float* scd1 = alloc(64); float* shd1 = alloc(64);
  float* scd2 = alloc(32); float* shd2 = alloc(32);
  float* smean = alloc(BB); float* srstd = alloc(BB);
  // large intermediates (enc overwrites Hc in place; kbuf reuses Z3)
  float* Z1 = alloc((size_t)BS * 32);
  float* Z2 = alloc((size_t)BS * 64);
  float* Z3 = alloc((size_t)BS * 128);   // later reused as key buffer
  float* H3 = alloc((size_t)BS * 128);
  float* Hc = alloc((size_t)BS * DM);    // conv output, then enc in place
  float* obuf  = alloc((size_t)BB * DM);
  float* y1raw = alloc((size_t)BB * 128);
  float* yln   = alloc((size_t)BB * 128);
  float* zd1   = alloc((size_t)BB * 64);
  float* zd2   = alloc((size_t)BB * 32);
  float* kbuf = Z3;

  const int rowsBlocks = (BS + 255) / 256;
  const int headBlocks = (BB + 255) / 256;

  // ---- MLP1 (BN folded into the input of the next layer) ----
  affine_relu_linear_kernel<10, 32><<<rowsBlocks, 256, 0, stream>>>(
      x, nullptr, nullptr, w1, b1, Z1, BS);
  bn_stats_kernel<<<32, 256, 0, stream>>>(Z1, BS, 32, g1, be1, sc1, sh1);
  affine_relu_linear_kernel<32, 64><<<rowsBlocks, 256, 0, stream>>>(
      Z1, sc1, sh1, w2, b2, Z2, BS);
  bn_stats_kernel<<<64, 256, 0, stream>>>(Z2, BS, 64, g2, be2, sc2, sh2);
  affine_relu_linear_kernel<64, 128><<<rowsBlocks, 256, 0, stream>>>(
      Z2, sc2, sh2, w3, b3, Z3, BS);
  bn_stats_kernel<<<128, 256, 0, stream>>>(Z3, BS, 128, g3, be3, sc3, sh3);
  bn_relu_ln_kernel<<<BS, 128, 0, stream>>>(Z3, sc3, sh3, ln_in_g, ln_in_b, H3, BS);

  // ---- conv GEMM [BS,128]@[128,256]^T : WMMA ----
  {
    long tiles = ((long)BS / 16) * (DM / 16);
    wmma_gemm_bias_kernel<128><<<(int)((tiles + 7) / 8), 256, 0, stream>>>(
        H3, conv_w, conv_b, Hc, BS, DM);
  }

  // ---- per-sample LN((DM,S)) + pos encoding (enc in place) ----
  sample_stats_kernel<<<BB, 256, 0, stream>>>(Hc, smean, srstd);
  encode_kernel<<<BS, 256, 0, stream>>>(Hc, smean, srstd, ln2_g, ln2_b, doyp, maskp, pos);

  // ---- key GEMM [BS,256]@[256,128]^T : WMMA ----
  {
    long tiles = ((long)BS / 16) * (128 / 16);
    wmma_gemm_bias_kernel<256><<<(int)((tiles + 7) / 8), 256, 0, stream>>>(
        Hc, wk, bk, kbuf, BS, 128);
  }

  // ---- attention: softmax over S, value aggregation ----
  attention_kernel<<<BB, 512, 0, stream>>>(kbuf, Hc, Q, obuf);

  // ---- head MLP ----
  {
    long tiles = ((long)BB / 16) * (128 / 16);
    wmma_gemm_bias_kernel<256><<<(int)((tiles + 7) / 8), 256, 0, stream>>>(
        obuf, w_m2, b_m2, y1raw, BB, 128);
  }
  bn_stats_kernel<<<128, 256, 0, stream>>>(y1raw, BB, 128, g_m2, be_m2, scm2, shm2);
  bn_relu_ln_kernel<<<BB, 128, 0, stream>>>(y1raw, scm2, shm2, out_g, out_b, yln, BB);
  affine_relu_linear_kernel<128, 64><<<headBlocks, 256, 0, stream>>>(
      yln, nullptr, nullptr, wd1, bd1, zd1, BB);
  bn_stats_kernel<<<64, 256, 0, stream>>>(zd1, BB, 64, gd1, bed1, scd1, shd1);
  affine_relu_linear_kernel<64, 32><<<headBlocks, 256, 0, stream>>>(
      zd1, scd1, shd1, wd2, bd2, zd2, BB);
  bn_stats_kernel<<<32, 256, 0, stream>>>(zd2, BB, 32, gd2, bed2, scd2, shd2);
  affine_relu_linear_kernel<32, 9><<<headBlocks, 256, 0, stream>>>(
      zd2, scd2, shd2, wd3, bd3, (float*)d_out, BB);
}
